// _TableTransform_1159641170473
// MI455X (gfx1250) — matrix operand order repaired and checked by reading, not compiled, not run on hardware
//
#include <hip/hip_runtime.h>
#include <cstdint>
#include <cstddef>

// ---------------------------------------------------------------------------
// CDNA5 / gfx1250 implementation of the hetero-GAT reference.
// GEMMs run on v_wmma_f32_16x16x32_f16 (f32 accumulate) with f16 LDS tiles,
// b128 DS fragment loads and vectorized global tile loads; edge softmax and
// message aggregation are atomic segment ops; LoRA and attention-vector
// algebra is folded on-device so the large hd projection never materializes.
// ---------------------------------------------------------------------------

typedef _Float16 f16;
typedef __attribute__((ext_vector_type(16))) _Float16 v16h;
typedef __attribute__((ext_vector_type(8)))  _Float16 h8;
typedef __attribute__((ext_vector_type(4)))  _Float16 h4;
typedef __attribute__((ext_vector_type(8)))  float    v8f;
typedef __attribute__((ext_vector_type(4)))  float    f32x4;

#define NHEAD 8
#define TN 64
#define TM 64
#define TK 32
#define XS_STRIDE 40   // halves; rows 16B-aligned, conflict-free column access
#define WS_STRIDE 40

// ------------------------------ utility kernels ----------------------------

__device__ __forceinline__ void atomicMaxF(float* addr, float v) {
  int iv = __float_as_int(v);
  if (iv >= 0) atomicMax((int*)addr, iv);
  else         atomicMin((unsigned int*)addr, (unsigned int)iv);
}

__global__ void fill_f32(float* __restrict__ p, float v, size_t n) {
  size_t i = (size_t)blockIdx.x * blockDim.x + threadIdx.x;
  size_t st = (size_t)gridDim.x * blockDim.x;
  for (; i < n; i += st) p[i] = v;
}

__global__ void concat3_k(const float* __restrict__ a, int ka,
                          const float* __restrict__ b, int kb,
                          const float* __restrict__ c, int kc,
                          float* __restrict__ out, int N) {
  int K = ka + kb + kc;
  size_t total = (size_t)N * K;
  size_t i = (size_t)blockIdx.x * blockDim.x + threadIdx.x;
  if (i >= total) return;
  int n = (int)(i / K), k = (int)(i % K);
  float v;
  if (k < ka)            v = a[(size_t)n * ka + k];
  else if (k < ka + kb)  v = b[(size_t)n * kb + (k - ka)];
  else                   v = c[(size_t)n * kc + (k - ka - kb)];
  out[i] = v;
}

// W_eff[m,k] = W[m,k] + 2 * sum_r B[m,r]*A[r,k]   (written as f16 for WMMA)
__global__ void lora_fold_k(const float* __restrict__ W, const float* __restrict__ A,
                            const float* __restrict__ B, f16* __restrict__ Weff,
                            int M, int K, int R) {
  int i = blockIdx.x * blockDim.x + threadIdx.x;
  if (i >= M * K) return;
  int m = i / K, k = i % K;
  float acc = W[i];
  for (int r = 0; r < R; ++r) acc += 2.0f * B[m * R + r] * A[r * K + k];
  Weff[i] = (f16)acc;
}

__global__ void cvt_f16_k(const float* __restrict__ in, f16* __restrict__ out, int n) {
  int i = blockIdx.x * blockDim.x + threadIdx.x;
  if (i < n) out[i] = (f16)in[i];
}

// V[h,k] = sum_d W[h*D+d, k] * avec[h*D+d]   (folds attention vector into W)
__global__ void fold_attn_k(const float* __restrict__ W, const float* __restrict__ avec,
                            float* __restrict__ V, int K, int D) {
  int i = blockIdx.x * blockDim.x + threadIdx.x;
  if (i >= NHEAD * K) return;
  int h = i / K, k = i % K;
  float acc = 0.f;
  for (int d = 0; d < D; ++d) acc += W[(size_t)(h * D + d) * K + k] * avec[h * D + d];
  V[i] = acc;
}

// out[n,h] = sum_k X[n,k] * V[h,k]    (el / er, M=8 so plain VALU beats WMMA)
__global__ void rowdot8_k(const float* __restrict__ X, const float* __restrict__ V,
                          float* __restrict__ out, int N, int K) {
  int i = blockIdx.x * blockDim.x + threadIdx.x;
  if (i >= N * NHEAD) return;
  int n = i >> 3, h = i & 7;
  float acc = 0.f;
  const float* xr = X + (size_t)n * K;
  const float* vr = V + (size_t)h * K;
  for (int k = 0; k < K; ++k) acc += xr[k] * vr[k];
  out[i] = acc;
}

// ------------------------------ WMMA GEMM ----------------------------------
// Y(NxM) = X(NxK, f32) @ W(MxK, f16)^T ; 256 thr = 8 waves; 64x64 block tile;
// each wave: 16 rows x 32 cols = two v_wmma_f32_16x16x32_f16 per K-step.
// X is converted f32->f16 on the way into LDS; fragments load as ds b128.
template <typename OutT>
__global__ __launch_bounds__(256) void gemm_xwt(const float* __restrict__ X,
                                                const f16* __restrict__ W,
                                                OutT* __restrict__ Y,
                                                int N, int M, int K) {
  __shared__ f16 Xs[TN * XS_STRIDE];
  __shared__ f16 Wsh[TM * WS_STRIDE];
  const int tid  = threadIdx.x;
  const int lane = tid & 31;
  const int wave = tid >> 5;
  const int wr   = wave & 3;   // wave row   (4 x 16 rows)
  const int wc   = wave >> 2;  // wave col   (2 x 32 cols)
  const int hi   = lane >> 4;  // lane half
  const int lm   = lane & 15;
  const int rowBase = blockIdx.y * TN;
  const int colBase = blockIdx.x * TM;

  // fast-path eligibility (checked independently for X and W tiles)
  const bool xvec = ((K & 3) == 0) && (rowBase + TN <= N);
  const bool wvec = ((K & 7) == 0) && (colBase + TM <= M);

  v8f acc0 = {}; v8f acc1 = {};

  for (int k0 = 0; k0 < K; k0 += TK) {
    const bool kin = (k0 + TK <= K);
    if (xvec && kin) {
      // 64x32 floats = 512 float4 chunks; 2 per thread; cvt to f16 for LDS
#pragma unroll
      for (int it = 0; it < 2; ++it) {
        int idx = tid + it * 256;
        int r = idx >> 3, c = (idx & 7) << 2;
        f32x4 v = *(const f32x4*)(X + (size_t)(rowBase + r) * K + k0 + c);
        h4 hv = { (f16)v.x, (f16)v.y, (f16)v.z, (f16)v.w };
        *(h4*)(&Xs[r * XS_STRIDE + c]) = hv;
      }
    } else {
      for (int t = tid; t < TN * TK; t += 256) {
        int r = t >> 5, c = t & 31;
        int gr = rowBase + r, gc = k0 + c;
        Xs[r * XS_STRIDE + c] =
            (gr < N && gc < K) ? (f16)X[(size_t)gr * K + gc] : (f16)0.f;
      }
    }
    if (wvec && kin) {
      // 64x32 halves = 256 h8 chunks; 1 per thread (16B global loads)
      int r = tid >> 2, c = (tid & 3) << 3;
      *(h8*)(&Wsh[r * WS_STRIDE + c]) =
          *(const h8*)(W + (size_t)(colBase + r) * K + k0 + c);
    } else {
      for (int t = tid; t < TM * TK; t += 256) {
        int r = t >> 5, c = t & 31;
        int gr = colBase + r, gc = k0 + c;
        Wsh[r * WS_STRIDE + c] =
            (gr < M && gc < K) ? W[(size_t)gr * K + gc] : (f16)0.f;
      }
    }
    __syncthreads();

    // A fragment: lane's K interleave = two contiguous 8-half runs (ISA 7.12.2)
    const f16* xr = &Xs[(wr * 16 + lm) * XS_STRIDE];
    h8 a0 = *(const h8*)(xr + 8 * hi);
    h8 a1 = *(const h8*)(xr + 16 + 8 * hi);
    v16h a = __builtin_shufflevector(a0, a1, 0, 1, 2, 3, 4, 5, 6, 7,
                                     8, 9, 10, 11, 12, 13, 14, 15);
    // B fragment: same K interleave as A = two contiguous 8-half runs
    const f16* w0 = &Wsh[(wc * 32 + lm) * WS_STRIDE + 8 * hi];
    const f16* w1 = &Wsh[(wc * 32 + 16 + lm) * WS_STRIDE + 8 * hi];
    h8 b00 = *(const h8*)(w0);
    h8 b01 = *(const h8*)(w0 + 16);
    h8 b10 = *(const h8*)(w1);
    h8 b11 = *(const h8*)(w1 + 16);
    v16h b0 = __builtin_shufflevector(b00, b01, 0, 1, 2, 3, 4, 5, 6, 7,
                                      8, 9, 10, 11, 12, 13, 14, 15);
    v16h b1 = __builtin_shufflevector(b10, b11, 0, 1, 2, 3, 4, 5, 6, 7,
                                      8, 9, 10, 11, 12, 13, 14, 15);

    acc0 = __builtin_amdgcn_wmma_f32_16x16x32_f16(false, a, false, b0, (short)0,
                                                  acc0, false, false);
    acc1 = __builtin_amdgcn_wmma_f32_16x16x32_f16(false, a, false, b1, (short)0,
                                                  acc1, false, false);
    __syncthreads();
  }

  // C/D layout: lane -> column, VGPR v -> row (v + 8*hi)
#pragma unroll
  for (int v = 0; v < 8; ++v) {
    int n  = rowBase + wr * 16 + v + hi * 8;
    int m0 = colBase + wc * 32 + lm;
    int m1 = m0 + 16;
    if (n < N) {
      if (m0 < M) Y[(size_t)n * M + m0] = (OutT)acc0[v];
      if (m1 < M) Y[(size_t)n * M + m1] = (OutT)acc1[v];
    }
  }
}

// ------------------------------ GAT edge kernels ---------------------------

__global__ void edge_max_k(const int* __restrict__ src, const int* __restrict__ dst,
                           const float* __restrict__ el, const float* __restrict__ er,
                           float* __restrict__ ex, float* __restrict__ m, int E) {
  int i = blockIdx.x * blockDim.x + threadIdx.x;
  if (i >= E * NHEAD) return;
  int e = i >> 3, h = i & 7;
  float v = el[src[e] * NHEAD + h] + er[dst[e] * NHEAD + h];
  v = (v > 0.f) ? v : 0.2f * v;          // leaky_relu(0.2)
  ex[i] = v;
  atomicMaxF(&m[dst[e] * NHEAD + h], v);
}

__global__ void edge_exp_k(const int* __restrict__ dst, float* __restrict__ ex,
                           const float* __restrict__ m, float* __restrict__ s, int E) {
  int i = blockIdx.x * blockDim.x + threadIdx.x;
  if (i >= E * NHEAD) return;
  int e = i >> 3, h = i & 7;
  float v = __expf(ex[i] - m[dst[e] * NHEAD + h]);
  ex[i] = v;
  atomicAdd(&s[dst[e] * NHEAD + h], v);
}

// one edge per blockIdx.x; src/dst uniform per block -> scalar loads
__global__ void edge_msg_k(const int* __restrict__ src, const int* __restrict__ dst,
                           const f16* __restrict__ hs, const float* __restrict__ ex,
                           const float* __restrict__ s, float* __restrict__ rst,
                           int HD, int dshift) {
  int e = blockIdx.x;
  int j = blockIdx.y * blockDim.x + threadIdx.x;
  if (j >= HD) return;
  int sv = src[e], dv = dst[e];
  int h = j >> dshift;
  float a = ex[e * NHEAD + h] / s[dv * NHEAD + h];
  atomicAdd(&rst[(size_t)dv * HD + j], (float)hs[(size_t)sv * HD + j] * a);
}

__global__ void gat_finalize_k(const float* __restrict__ rst, const float* __restrict__ dstf,
                               float* __restrict__ out, int Nd, int D, int dshift) {
  int i = blockIdx.x * blockDim.x + threadIdx.x;
  if (i >= Nd * D) return;
  int n = i >> dshift, d = i & (D - 1);
  float acc = 0.f;
  for (int h = 0; h < NHEAD; ++h) acc += rst[(size_t)n * NHEAD * D + h * D + d];
  out[i] = acc * 0.125f + dstf[i];
}

__global__ void avg2_k(const float* __restrict__ a, const float* __restrict__ b,
                       float* __restrict__ out, size_t n) {
  size_t i = (size_t)blockIdx.x * blockDim.x + threadIdx.x;
  if (i < n) out[i] = 0.5f * (a[i] + b[i]);
}

// ------------------------------ host driver --------------------------------

static inline unsigned blocks1d(size_t n) { return (unsigned)((n + 255) / 256); }

extern "C" void kernel_launch(void* const* d_in, const int* in_sizes, int n_in,
                              void* d_out, int out_size, void* d_ws, size_t ws_size,
                              hipStream_t stream) {
  (void)n_in; (void)out_size; (void)ws_size;
  // ---- inputs --------------------------------------------------------------
  const float* table_features = (const float*)d_in[0];
  const float* table_onehot   = (const float*)d_in[1];
  const float* table_dense    = (const float*)d_in[2];
  const float* col_stats      = (const float*)d_in[3];
  const float* filt_feat      = (const float*)d_in[4];
  const float* join_feat      = (const float*)d_in[5];
  const int* c2t_src = (const int*)d_in[6];
  const int* c2t_dst = (const int*)d_in[7];
  const int* c2f_src = (const int*)d_in[8];
  const int* c2f_dst = (const int*)d_in[9];
  const int* c2j_src = (const int*)d_in[10];
  const int* c2j_dst = (const int*)d_in[11];

  const int nt = in_sizes[0] / 3;
  const int nc = in_sizes[3] / 64;
  const int nf = in_sizes[4];
  const int nj = in_sizes[5];
  const int Ec2t = in_sizes[6];
  const int Ec2f = in_sizes[8];
  const int Ec2j = in_sizes[10];

  // params flattened in dict order: 5 LoRA x {W,A,B}, 7 GAT x {Ws,Wd,al,ar}
  const float* P[48];
  for (int i = 12; i < 55; ++i) P[i - 12] = (const float*)d_in[i];
  const float **fc_table = &P[0], **fc_column = &P[3], **fc_filter = &P[6],
              **fc_join = &P[9], **fc_out = &P[12];
  const float **c2t1 = &P[15], **t2c1 = &P[19], **c2f1 = &P[23], **c2j1 = &P[27],
              **f2c1 = &P[31], **j2c1 = &P[35], **c2t2 = &P[39];

  // ---- workspace carve -----------------------------------------------------
  char* wp = (char*)d_ws;
  auto carve = [&](size_t bytes) -> void* {
    void* r = (void*)wp; wp += (bytes + 255) & ~(size_t)255; return r;
  };
  float* te   = (float*)carve((size_t)nt * 64 * 4);
  float* ce   = (float*)carve((size_t)nc * 32 * 4);
  float* fe   = (float*)carve((size_t)nf * 16 * 4);
  float* je   = (float*)carve((size_t)nj * 16 * 4);
  float* t1   = (float*)carve((size_t)nt * 64 * 4);
  float* c1   = (float*)carve((size_t)nc * 32 * 4);
  float* f1   = (float*)carve((size_t)nf * 16 * 4);
  float* j1   = (float*)carve((size_t)nj * 16 * 4);
  float* c2fb = (float*)carve((size_t)nc * 32 * 4);
  float* c2jb = (float*)carve((size_t)nc * 32 * 4);
  float* cavg = (float*)carve((size_t)nc * 32 * 4);
  float* t2   = (float*)carve((size_t)nt * 64 * 4);
  float* tcat = (float*)carve((size_t)nt * 56 * 4);
  float* ocat = (float*)carve((size_t)nt * 67 * 4);
  f16*   hs   = (f16*)  carve((size_t)nc * 512 * 2);   // max Ns*H*D (f16)
  float* rst  = (float*)carve((size_t)nc * 256 * 4);   // max Nd*H*D
  float* el   = (float*)carve((size_t)nc * 8 * 4);
  float* er   = (float*)carve((size_t)nc * 8 * 4);
  float* mbuf = (float*)carve((size_t)nc * 8 * 4);
  float* sbuf = (float*)carve((size_t)nc * 8 * 4);
  float* exb  = (float*)carve((size_t)Ec2t * 8 * 4);   // max E*8, e then ex
  f16*   wscr = (f16*)  carve((size_t)512 * 64 * 2);   // f16 weight staging
  float* Vs   = (float*)carve((size_t)8 * 64 * 4);
  float* Vd   = (float*)carve((size_t)8 * 64 * 4);

  // ---- launch helpers ------------------------------------------------------
  auto gemm_f32 = [&](const float* X, const f16* W, float* Y, int N, int M, int K) {
    dim3 g((M + TM - 1) / TM, (N + TN - 1) / TN);
    gemm_xwt<float><<<g, 256, 0, stream>>>(X, W, Y, N, M, K);
  };
  auto gemm_f16 = [&](const float* X, const f16* W, f16* Y, int N, int M, int K) {
    dim3 g((M + TM - 1) / TM, (N + TN - 1) / TN);
    gemm_xwt<f16><<<g, 256, 0, stream>>>(X, W, Y, N, M, K);
  };
  auto lora = [&](const float** p, const float* X, float* Y, int N, int M, int K) {
    lora_fold_k<<<blocks1d((size_t)M * K), 256, 0, stream>>>(p[0], p[1], p[2], wscr, M, K, 8);
    gemm_f32(X, wscr, Y, N, M, K);
  };
  auto run_gat = [&](const float* srcf, const float* dstf, int Ns, int Nd,
                     int Ks, int Kd, int D, const int* si, const int* di, int E,
                     const float** gp, float* out) {
    const float *Ws = gp[0], *Wd = gp[1], *al = gp[2], *ar = gp[3];
    const int HD = NHEAD * D;
    const int dshift = (D == 16) ? 4 : ((D == 32) ? 5 : 6);
    cvt_f16_k  <<<blocks1d((size_t)HD * Ks), 256, 0, stream>>>(Ws, wscr, HD * Ks);
    fold_attn_k<<<blocks1d((size_t)NHEAD * Ks), 256, 0, stream>>>(Ws, al, Vs, Ks, D);
    fold_attn_k<<<blocks1d((size_t)NHEAD * Kd), 256, 0, stream>>>(Wd, ar, Vd, Kd, D);
    gemm_f16(srcf, wscr, hs, Ns, HD, Ks);                         // hs (f16)
    rowdot8_k<<<blocks1d((size_t)Ns * 8), 256, 0, stream>>>(srcf, Vs, el, Ns, Ks);
    rowdot8_k<<<blocks1d((size_t)Nd * 8), 256, 0, stream>>>(dstf, Vd, er, Nd, Kd);
    fill_f32<<<blocks1d((size_t)Nd * 8), 256, 0, stream>>>(mbuf, -1e30f, (size_t)Nd * 8);
    fill_f32<<<blocks1d((size_t)Nd * 8), 256, 0, stream>>>(sbuf, 0.f, (size_t)Nd * 8);
    fill_f32<<<blocks1d((size_t)Nd * HD), 256, 0, stream>>>(rst, 0.f, (size_t)Nd * HD);
    edge_max_k<<<blocks1d((size_t)E * 8), 256, 0, stream>>>(si, di, el, er, exb, mbuf, E);
    edge_exp_k<<<blocks1d((size_t)E * 8), 256, 0, stream>>>(di, exb, mbuf, sbuf, E);
    dim3 mg((unsigned)E, (unsigned)((HD + 255) / 256));
    int mb = (HD < 256) ? HD : 256;
    edge_msg_k<<<mg, mb, 0, stream>>>(si, di, hs, exb, sbuf, rst, HD, dshift);
    gat_finalize_k<<<blocks1d((size_t)Nd * D), 256, 0, stream>>>(rst, dstf, out, Nd, D, dshift);
  };

  // ---- phase A: node-type embeddings --------------------------------------
  concat3_k<<<blocks1d((size_t)nt * 56), 256, 0, stream>>>(
      table_features, 3, table_onehot, 21, table_dense, 32, tcat, nt);
  lora(fc_table,  tcat,      te, nt, 64, 56);
  lora(fc_column, col_stats, ce, nc, 32, 64);
  lora(fc_filter, filt_feat, fe, nf, 16, 1);
  lora(fc_join,   join_feat, je, nj, 16, 1);

  // ---- phase B: heterogeneous message passing ------------------------------
  run_gat(ce, te, nc, nt, 32, 64, 64, c2t_src, c2t_dst, Ec2t, c2t1, t1);
  run_gat(t1, ce, nt, nc, 64, 32, 32, c2t_dst, c2t_src, Ec2t, t2c1, c1);
  run_gat(c1, fe, nc, nf, 32, 16, 16, c2f_src, c2f_dst, Ec2f, c2f1, f1);
  run_gat(c1, je, nc, nj, 32, 16, 16, c2j_src, c2j_dst, Ec2j, c2j1, j1);
  run_gat(f1, c1, nf, nc, 16, 32, 32, c2f_dst, c2f_src, Ec2f, f2c1, c2fb);
  run_gat(j1, c1, nj, nc, 16, 32, 32, c2j_dst, c2j_src, Ec2j, j2c1, c2jb);
  avg2_k<<<blocks1d((size_t)nc * 32), 256, 0, stream>>>(c2fb, c2jb, cavg, (size_t)nc * 32);
  run_gat(cavg, t1, nc, nt, 32, 64, 64, c2t_src, c2t_dst, Ec2t, c2t2, t2);

  // ---- phase C: output head ------------------------------------------------
  concat3_k<<<blocks1d((size_t)nt * 67), 256, 0, stream>>>(
      t2, 64, table_features, 3, table_features, 0, ocat, nt);
  lora(fc_out, ocat, (float*)d_out, nt, 128, 67);
}
